// MySpikeGPT_51960514347666
// MI455X (gfx1250) — matrix operand order repaired and checked by reading, not compiled, not run on hardware
//
#include <hip/hip_runtime.h>
#include <hip/hip_bf16.h>

// ---------------------------------------------------------------------------
// MySpikeGPT forward for MI455X (gfx1250).
// GEMMs: v_wmma_f32_16x16x32_bf16 with double-buffered LDS staging via
// global_load_async_to_lds_b128 (ASYNCcnt) so DMA overlaps the WMMAs.
// Weights are pre-transposed (N-major, N padded to x128) so both GEMM tiles
// stage as contiguous 16B chunks with zero guards in the inner loop.
// ---------------------------------------------------------------------------

#define VOCAB 50257
#define DMODEL 768
#define CTXLEN 1024
#define FFDIM 3072
#define NLAYER 12
#define TSTEPS 4
#define BATCH 2
#define MROWS (TSTEPS * BATCH * CTXLEN)   // 8192
#define PROWS (BATCH * CTXLEN)            // 2048

typedef __bf16 bf16_t;
typedef __attribute__((ext_vector_type(16))) __bf16 v16bf;
typedef __attribute__((ext_vector_type(8)))  __bf16 bf16x8;
typedef __attribute__((ext_vector_type(8)))  float  v8f;

#if __has_builtin(__builtin_amdgcn_global_load_async_to_lds_b128)
#define HAVE_ASYNC_LDS 1
#endif

// Builtin parameter types (from hipcc diagnostic): int4 with addrspace(1)/(3).
typedef int intx4 __attribute__((vector_size(4 * sizeof(int))));
typedef __attribute__((address_space(1))) intx4 as1_intx4;
typedef __attribute__((address_space(3))) intx4 as3_intx4;

// Generic->LDS pointer: low 32 bits of a generic LDS address are the LDS byte
// offset (aperture base has zero low half), so integer truncation is exact.
__device__ __forceinline__ as3_intx4* lds_cast(void* p) {
  return (as3_intx4*)(unsigned int)(unsigned long long)p;
}
__device__ __forceinline__ as1_intx4* glb_cast(const void* p) {
  return (as1_intx4*)(unsigned long long)p;
}

__device__ __forceinline__ void wait_async0() {
#if defined(HAVE_ASYNC_LDS)
#if __has_builtin(__builtin_amdgcn_s_wait_asynccnt)
  __builtin_amdgcn_s_wait_asynccnt(0);
#else
  asm volatile("s_wait_asynccnt 0x0" ::: "memory");
#endif
#endif
}

__device__ __forceinline__ bf16_t f2bf(float f) {
  union { float f; unsigned u; } v; v.f = f;
  unsigned r = v.u + 0x7FFFu + ((v.u >> 16) & 1u);   // round-to-nearest-even
  unsigned short h = (unsigned short)(r >> 16);
  bf16_t out; __builtin_memcpy(&out, &h, 2);
  return out;
}

// ---------------------------------------------------------------------------
// Elementwise bf16 converter (activations, row-major A matrices)
// ---------------------------------------------------------------------------
__global__ void relu_bf16_kernel(const float* __restrict__ x, bf16_t* __restrict__ y, long n) {
  for (long i = (long)blockIdx.x * blockDim.x + threadIdx.x; i < n;
       i += (long)gridDim.x * blockDim.x)
    y[i] = f2bf(fmaxf(x[i], 0.f));
}

// ---------------------------------------------------------------------------
// Weight staging: Wt[n][k] = bf16(W[k][n]); rows n in [N, Npad) zero-filled.
// 32x32 tiles through LDS; 256 threads (32x8).
// ---------------------------------------------------------------------------
__global__ __launch_bounds__(256) void cvt_transpose_bf16_kernel(
    const float* __restrict__ w, bf16_t* __restrict__ wt, int K, int N, int Npad) {
  __shared__ float tile[32][33];
  const int nb = blockIdx.x * 32;
  const int kb = blockIdx.y * 32;
  const int tx = threadIdx.x & 31;
  const int ty = threadIdx.x >> 5;           // 0..7
  #pragma unroll
  for (int i = ty; i < 32; i += 8) {
    int n = nb + tx;
    tile[i][tx] = (n < N) ? w[(size_t)(kb + i) * N + n] : 0.f;
  }
  __syncthreads();
  #pragma unroll
  for (int i = ty; i < 32; i += 8) {
    int n = nb + i;
    if (n < Npad) wt[(size_t)n * K + kb + tx] = f2bf(tile[tx][i]);
  }
}

// ---------------------------------------------------------------------------
// Embedding gather + sinusoidal position encoding + LayerNorm. One block/row.
// ---------------------------------------------------------------------------
__global__ __launch_bounds__(256) void embed_ln_kernel(
    const int* __restrict__ tokens, const float* __restrict__ emb,
    const float* __restrict__ g, const float* __restrict__ b,
    float* __restrict__ out) {
  const int row = blockIdx.x;                    // [0, MROWS)
  const int s   = row % CTXLEN;
  const int tok = tokens[row];
  const float* e = emb + (size_t)tok * DMODEL;

  float xv[3];                                   // DMODEL / 256 = 3
  float acc = 0.f;
  #pragma unroll
  for (int i = 0; i < 3; i++) {
    int j = threadIdx.x + i * 256;
    float expo = (float)(j & ~1) * (1.0f / DMODEL);
    float ang  = (float)s * __expf(expo * -9.210340371976184f); // /10000^expo
    float pe   = (j & 1) ? __cosf(ang) : __sinf(ang);
    xv[i] = e[j] + pe;
    acc += xv[i];
  }
  __shared__ float red[256];
  red[threadIdx.x] = acc; __syncthreads();
  for (int st = 128; st > 0; st >>= 1) {
    if (threadIdx.x < st) red[threadIdx.x] += red[threadIdx.x + st];
    __syncthreads();
  }
  float mean = red[0] * (1.0f / DMODEL);
  __syncthreads();
  acc = 0.f;
  #pragma unroll
  for (int i = 0; i < 3; i++) { float d = xv[i] - mean; acc += d * d; }
  red[threadIdx.x] = acc; __syncthreads();
  for (int st = 128; st > 0; st >>= 1) {
    if (threadIdx.x < st) red[threadIdx.x] += red[threadIdx.x + st];
    __syncthreads();
  }
  float rstd = rsqrtf(red[0] * (1.0f / DMODEL) + 1e-5f);
  #pragma unroll
  for (int i = 0; i < 3; i++) {
    int j = threadIdx.x + i * 256;
    out[(size_t)row * DMODEL + j] = (xv[i] - mean) * rstd * g[j] + b[j];
  }
}

// ---------------------------------------------------------------------------
// Row-wise LayerNorm epilogue: optional relu, residual add, f32/bf16 outputs.
// ---------------------------------------------------------------------------
__global__ __launch_bounds__(256) void ln_kernel(
    const float* __restrict__ x, const float* __restrict__ g, const float* __restrict__ b,
    const float* __restrict__ resid, float* __restrict__ outf, bf16_t* __restrict__ outb,
    int cols, int relu) {
  const int row = blockIdx.x;
  const float* xr = x + (size_t)row * cols;
  __shared__ float red[256];
  float s = 0.f;
  for (int c = threadIdx.x; c < cols; c += 256) s += xr[c];
  red[threadIdx.x] = s; __syncthreads();
  for (int st = 128; st > 0; st >>= 1) {
    if (threadIdx.x < st) red[threadIdx.x] += red[threadIdx.x + st];
    __syncthreads();
  }
  float mean = red[0] / (float)cols;
  __syncthreads();
  float v = 0.f;
  for (int c = threadIdx.x; c < cols; c += 256) { float d = xr[c] - mean; v += d * d; }
  red[threadIdx.x] = v; __syncthreads();
  for (int st = 128; st > 0; st >>= 1) {
    if (threadIdx.x < st) red[threadIdx.x] += red[threadIdx.x + st];
    __syncthreads();
  }
  float rstd = rsqrtf(red[0] / (float)cols + 1e-5f);
  for (int c = threadIdx.x; c < cols; c += 256) {
    float y = (xr[c] - mean) * rstd * g[c] + b[c];
    if (relu) y = fmaxf(y, 0.f);
    if (resid) y += resid[(size_t)row * cols + c];
    if (outf) outf[(size_t)row * cols + c] = y;
    if (outb) outb[(size_t)row * cols + c] = f2bf(y);
  }
}

// ---------------------------------------------------------------------------
// Spiking linear attention pieces.
// ---------------------------------------------------------------------------
__global__ void qksum_kernel(const float* __restrict__ Q, const float* __restrict__ K,
                             float* __restrict__ qk) {
  int i = blockIdx.x * blockDim.x + threadIdx.x;     // [0, T*B*D)
  if (i >= TSTEPS * BATCH * DMODEL) return;
  int tb = i / DMODEL, c = i % DMODEL;
  const float* q = Q + (size_t)tb * CTXLEN * DMODEL + c;
  const float* k = K + (size_t)tb * CTXLEN * DMODEL + c;
  float s = 0.f;
  for (int t = 0; t < CTXLEN; t++) s += q[(size_t)t * DMODEL] * k[(size_t)t * DMODEL];
  qk[i] = fmaxf(s, 0.f);
}

__global__ void attmul_kernel(const float* __restrict__ Vv, const float* __restrict__ qk,
                              bf16_t* __restrict__ out) {
  const long n = (long)MROWS * DMODEL;
  for (long i = (long)blockIdx.x * blockDim.x + threadIdx.x; i < n;
       i += (long)gridDim.x * blockDim.x) {
    long tb = i / ((long)CTXLEN * DMODEL);
    int  c  = (int)(i % DMODEL);
    out[i] = f2bf(Vv[i] * qk[tb * DMODEL + c]);
  }
}

__global__ void pool_kernel(const float* __restrict__ h, bf16_t* __restrict__ out) {
  const long n = (long)PROWS * DMODEL;
  for (long i = (long)blockIdx.x * blockDim.x + threadIdx.x; i < n;
       i += (long)gridDim.x * blockDim.x) {
    float s = 0.f;
    #pragma unroll
    for (int t = 0; t < TSTEPS; t++) s += fmaxf(h[(size_t)t * n + i], 0.f);
    out[i] = f2bf(s * (1.0f / TSTEPS));
  }
}

// ---------------------------------------------------------------------------
// bf16 WMMA GEMM: C[M,N] = A[M,K] * Bt[N][K]^T.
//   A: bf16 row-major [M,K]; Bt: bf16 N-major [Npad,K] (pre-transposed, padded)
// Block tile 128x128x32, 8 waves (2x4), wave tile 64x32 = 8 accumulators.
// Double-buffered LDS staging via async global->LDS DMA: per K-step,
//   wait(cur DMA) -> barrier -> issue DMA(next tile, other buffer) -> compute.
// One barrier per K-step; DMA overlaps WMMA. Requires M%128==0, K%32==0,
// Npad%128==0 (true at all call sites).
// ---------------------------------------------------------------------------
#define BM 128
#define BN 128
#define BK 32
#define LDSS 40   // padded LDS row stride (elements); 80B rows, 16B aligned

__global__ __launch_bounds__(256) void gemm_bf16_wmma(
    const bf16_t* __restrict__ A, const bf16_t* __restrict__ Bt,
    float* __restrict__ C, int M, int N, int K) {
  __shared__ bf16_t Al[2][BM * LDSS];
  __shared__ bf16_t Bl[2][BN * LDSS];

  const int tid  = threadIdx.x;
  const int lane = tid & 31;
  const int wave = tid >> 5;     // 0..7
  const int wm   = wave >> 2;    // 0..1  (64 rows each)
  const int wn   = wave & 3;     // 0..3  (32 cols each)
  const int bm   = blockIdx.y * BM;
  const int bn   = blockIdx.x * BN;
  const int lane_r  = lane & 15;
  const int lane_hi = lane >> 4;

  // per-thread staging chunk: rows r0 and r0+64, 16B column chunk c8
  const int r0 = tid >> 2;             // 0..63
  const int c8 = (tid & 3) * 8;        // 0,8,16,24

  // global staging base pointers (k advances by BK per tile)
  const bf16_t* ga0 = A  + (size_t)(bm + r0)      * K + c8;
  const bf16_t* ga1 = A  + (size_t)(bm + r0 + 64) * K + c8;
  const bf16_t* gb0 = Bt + (size_t)(bn + r0)      * K + c8;
  const bf16_t* gb1 = Bt + (size_t)(bn + r0 + 64) * K + c8;

  v8f acc[4][2];
  #pragma unroll
  for (int i = 0; i < 4; i++)
    #pragma unroll
    for (int j = 0; j < 2; j++)
      #pragma unroll
      for (int e = 0; e < 8; e++) acc[i][j][e] = 0.f;

  const int nk = K / BK;

  auto stage = [&](int k0, int buf) {
    bf16_t* la0 = &Al[buf][r0 * LDSS + c8];
    bf16_t* la1 = &Al[buf][(r0 + 64) * LDSS + c8];
    bf16_t* lb0 = &Bl[buf][r0 * LDSS + c8];
    bf16_t* lb1 = &Bl[buf][(r0 + 64) * LDSS + c8];
#if defined(HAVE_ASYNC_LDS)
    __builtin_amdgcn_global_load_async_to_lds_b128(glb_cast(ga0 + k0), lds_cast(la0), 0, 0);
    __builtin_amdgcn_global_load_async_to_lds_b128(glb_cast(ga1 + k0), lds_cast(la1), 0, 0);
    __builtin_amdgcn_global_load_async_to_lds_b128(glb_cast(gb0 + k0), lds_cast(lb0), 0, 0);
    __builtin_amdgcn_global_load_async_to_lds_b128(glb_cast(gb1 + k0), lds_cast(lb1), 0, 0);
#else
    *(bf16x8*)la0 = *(const bf16x8*)(ga0 + k0);
    *(bf16x8*)la1 = *(const bf16x8*)(ga1 + k0);
    *(bf16x8*)lb0 = *(const bf16x8*)(gb0 + k0);
    *(bf16x8*)lb1 = *(const bf16x8*)(gb1 + k0);
#endif
  };

  // prologue: stage tile 0 into buffer 0
  stage(0, 0);

  for (int kt = 0; kt < nk; kt++) {
    const int cur = kt & 1;
    wait_async0();     // this wave's DMA for buffer `cur` has landed in LDS
    __syncthreads();   // all waves' DMA landed; all waves done reading `cur^1`
    if (kt + 1 < nk) {
      stage((kt + 1) * BK, cur ^ 1);   // DMA next tile, overlaps compute below
      // warm L2 two tiles ahead of consumption (one ahead of the DMA)
      if (kt + 2 < nk) {
        __builtin_prefetch(ga0 + (kt + 2) * BK, 0, 0);
        __builtin_prefetch(gb0 + (kt + 2) * BK, 0, 0);
      }
    }

    // ---- fragments from buffer `cur`
    // A 16x32 bf16: lane(0-15)=row M, K halves kb and kb+16 (kb = 8*lane_hi)
    v16bf afr[4];
    #pragma unroll
    for (int tm = 0; tm < 4; tm++) {
      int row = wm * 64 + tm * 16 + lane_r;
      int kb  = lane_hi * 8;
      bf16x8 lo = *(const bf16x8*)&Al[cur][row * LDSS + kb];
      bf16x8 hi = *(const bf16x8*)&Al[cur][row * LDSS + kb + 16];
      #pragma unroll
      for (int e = 0; e < 8; e++) { afr[tm][e] = lo[e]; afr[tm][e + 8] = hi[e]; }
    }
    // B 32x16 bf16: lane_r = col N, K halves kb and kb+16 (kb = 8*lane_hi)
    v16bf bfr[2];
    #pragma unroll
    for (int tn = 0; tn < 2; tn++) {
      int col = wn * 32 + tn * 16 + lane_r;
      int kb  = lane_hi * 8;
      bf16x8 lo = *(const bf16x8*)&Bl[cur][col * LDSS + kb];
      bf16x8 hi = *(const bf16x8*)&Bl[cur][col * LDSS + kb + 16];
      #pragma unroll
      for (int e = 0; e < 8; e++) { bfr[tn][e] = lo[e]; bfr[tn][e + 8] = hi[e]; }
    }
    // ---- 8 WMMAs per wave per K-step
    #pragma unroll
    for (int tm = 0; tm < 4; tm++)
      #pragma unroll
      for (int tn = 0; tn < 2; tn++)
        acc[tm][tn] = __builtin_amdgcn_wmma_f32_16x16x32_bf16(
            false, afr[tm], false, bfr[tn], (short)0, acc[tm][tn], false, false);
  }

  // ---- store: VGPR v -> row (v + 8*lane_hi), col = lane_r within 16x16 tile
  #pragma unroll
  for (int tm = 0; tm < 4; tm++) {
    int grow = bm + wm * 64 + tm * 16 + lane_hi * 8;
    #pragma unroll
    for (int tn = 0; tn < 2; tn++) {
      int gcol = bn + wn * 32 + tn * 16 + lane_r;
      if (gcol < N) {
        #pragma unroll
        for (int v = 0; v < 8; v++)
          C[(size_t)(grow + v) * N + gcol] = acc[tm][tn][v];
      }
    }
  }
}

// ---------------------------------------------------------------------------
// Host orchestration
// ---------------------------------------------------------------------------
static inline unsigned cdivu(unsigned a, unsigned b) { return (a + b - 1) / b; }

extern "C" void kernel_launch(void* const* d_in, const int* in_sizes, int n_in,
                              void* d_out, int out_size, void* d_ws, size_t ws_size,
                              hipStream_t stream) {
  const int*   tokens = (const int*)  d_in[0];
  const float* emb    = (const float*)d_in[1];
  const float* enc_g  = (const float*)d_in[2];
  const float* enc_b  = (const float*)d_in[3];
  const float* wq     = (const float*)d_in[4];
  const float* wk     = (const float*)d_in[5];
  const float* wv     = (const float*)d_in[6];
  const float* wo     = (const float*)d_in[7];
  const float* lnq_g  = (const float*)d_in[8];
  const float* lnq_b  = (const float*)d_in[9];
  const float* lnk_g  = (const float*)d_in[10];
  const float* lnk_b  = (const float*)d_in[11];
  const float* lnv_g  = (const float*)d_in[12];
  const float* lnv_b  = (const float*)d_in[13];
  const float* lno_g  = (const float*)d_in[14];
  const float* lno_b  = (const float*)d_in[15];
  const float* w1     = (const float*)d_in[16];
  const float* ln1_g  = (const float*)d_in[17];
  const float* ln1_b  = (const float*)d_in[18];
  const float* w2     = (const float*)d_in[19];
  const float* ln2_g  = (const float*)d_in[20];
  const float* ln2_b  = (const float*)d_in[21];
  const float* w_out  = (const float*)d_in[22];
  float* out = (float*)d_out;

  const unsigned VPAD = cdivu(VOCAB, 128) * 128;   // 50304

  // workspace carve-out (256B aligned slabs)
  char* ws = (char*)d_ws;
  size_t off = 0;
  auto take = [&](size_t bytes) -> void* {
    void* p = ws + off;
    off = (off + bytes + 255) & ~(size_t)255;
    return p;
  };
  float*  h    = (float*) take((size_t)MROWS * DMODEL * 4);       // residual stream
  float*  go   = (float*) take((size_t)MROWS * FFDIM  * 4);       // GEMM out (max)
  bf16_t* act  = (bf16_t*)take((size_t)MROWS * FFDIM  * 2);       // GEMM A input (max)
  bf16_t* wbuf = (bf16_t*)take((size_t)VPAD  * DMODEL * 2);       // Wt staging (max)
  float*  Qf   = (float*) take((size_t)MROWS * DMODEL * 4);
  float*  Kf   = (float*) take((size_t)MROWS * DMODEL * 4);
  float*  Vf   = (float*) take((size_t)MROWS * DMODEL * 4);
  float*  qk   = (float*) take((size_t)TSTEPS * BATCH * DMODEL * 4);

  const dim3 blk(256);
  const int EWGRID = 2048;
  const dim3 tgridDD(DMODEL / 32, DMODEL / 32);       // W[D,D] transpose
  const dim3 ggridD(DMODEL / BN, MROWS / BM);         // C[8192, 768]

  // 1) h = LN(emb[tokens] + poe)
  embed_ln_kernel<<<MROWS, blk, 0, stream>>>(tokens, emb, enc_g, enc_b, h);

  for (int l = 0; l < NLAYER; l++) {
    const size_t wsq = (size_t)l * DMODEL * DMODEL;
    // t = relu(h)  (bf16 GEMM A input)
    relu_bf16_kernel<<<EWGRID, blk, 0, stream>>>(h, act, (long)MROWS * DMODEL);

    // Q = relu(LN(t @ wq))
    cvt_transpose_bf16_kernel<<<tgridDD, blk, 0, stream>>>(wq + wsq, wbuf, DMODEL, DMODEL, DMODEL);
    gemm_bf16_wmma<<<ggridD, blk, 0, stream>>>(act, wbuf, go, MROWS, DMODEL, DMODEL);
    ln_kernel<<<MROWS, blk, 0, stream>>>(go, lnq_g + l * DMODEL, lnq_b + l * DMODEL,
                                         nullptr, Qf, nullptr, DMODEL, 1);
    // K
    cvt_transpose_bf16_kernel<<<tgridDD, blk, 0, stream>>>(wk + wsq, wbuf, DMODEL, DMODEL, DMODEL);
    gemm_bf16_wmma<<<ggridD, blk, 0, stream>>>(act, wbuf, go, MROWS, DMODEL, DMODEL);
    ln_kernel<<<MROWS, blk, 0, stream>>>(go, lnk_g + l * DMODEL, lnk_b + l * DMODEL,
                                         nullptr, Kf, nullptr, DMODEL, 1);
    // V
    cvt_transpose_bf16_kernel<<<tgridDD, blk, 0, stream>>>(wv + wsq, wbuf, DMODEL, DMODEL, DMODEL);
    gemm_bf16_wmma<<<ggridD, blk, 0, stream>>>(act, wbuf, go, MROWS, DMODEL, DMODEL);
    ln_kernel<<<MROWS, blk, 0, stream>>>(go, lnv_g + l * DMODEL, lnv_b + l * DMODEL,
                                         nullptr, Vf, nullptr, DMODEL, 1);

    // QK = relu(sum_s Q*K); att = V * QK (bf16 GEMM A input)
    qksum_kernel<<<cdivu(TSTEPS * BATCH * DMODEL, 256), blk, 0, stream>>>(Qf, Kf, qk);
    attmul_kernel<<<EWGRID, blk, 0, stream>>>(Vf, qk, act);

    // h = h + LN(att @ wo)
    cvt_transpose_bf16_kernel<<<tgridDD, blk, 0, stream>>>(wo + wsq, wbuf, DMODEL, DMODEL, DMODEL);
    gemm_bf16_wmma<<<ggridD, blk, 0, stream>>>(act, wbuf, go, MROWS, DMODEL, DMODEL);
    ln_kernel<<<MROWS, blk, 0, stream>>>(go, lno_g + l * DMODEL, lno_b + l * DMODEL,
                                         h, h, nullptr, DMODEL, 0);

    // FFN: f = relu(h); f = relu(LN(f @ w1)); h = h + LN(f @ w2)
    relu_bf16_kernel<<<EWGRID, blk, 0, stream>>>(h, act, (long)MROWS * DMODEL);
    cvt_transpose_bf16_kernel<<<dim3(FFDIM / 32, DMODEL / 32), blk, 0, stream>>>(
        w1 + (size_t)l * DMODEL * FFDIM, wbuf, DMODEL, FFDIM, FFDIM);
    gemm_bf16_wmma<<<dim3(FFDIM / BN, MROWS / BM), blk, 0, stream>>>(
        act, wbuf, go, MROWS, FFDIM, DMODEL);
    ln_kernel<<<MROWS, blk, 0, stream>>>(go, ln1_g + l * FFDIM, ln1_b + l * FFDIM,
                                         nullptr, nullptr, act, FFDIM, 1);
    cvt_transpose_bf16_kernel<<<dim3(DMODEL / 32, FFDIM / 32), blk, 0, stream>>>(
        w2 + (size_t)l * FFDIM * DMODEL, wbuf, FFDIM, DMODEL, DMODEL);
    gemm_bf16_wmma<<<ggridD, blk, 0, stream>>>(act, wbuf, go, MROWS, DMODEL, FFDIM);
    ln_kernel<<<MROWS, blk, 0, stream>>>(go, ln2_g + l * DMODEL, ln2_b + l * DMODEL,
                                         h, h, nullptr, DMODEL, 0);
  }

  // pooled = mean_t relu(h) -> bf16 ; out = pooled @ w_out
  pool_kernel<<<EWGRID, blk, 0, stream>>>(h, act);
  cvt_transpose_bf16_kernel<<<dim3(VPAD / 32, DMODEL / 32), blk, 0, stream>>>(
      w_out, wbuf, DMODEL, VOCAB, VPAD);
  gemm_bf16_wmma<<<dim3(VPAD / BN, PROWS / BM), blk, 0, stream>>>(
      act, wbuf, out, PROWS, VOCAB, DMODEL);
}
